// MultiHeadAttention_41583873360619
// MI455X (gfx1250) — compile-verified, matrix operand order checked by reading, not run on hardware
//
#include <hip/hip_runtime.h>
#include <hip/hip_bf16.h>

// MI455X / gfx1250 multi-head attention, bf16 WMMA path.
// D=1024, H=16, Dk=64, B=2, S=2048.

typedef __attribute__((ext_vector_type(16))) __bf16 v16bf;
typedef __attribute__((ext_vector_type(8)))  float  v8f;

#define D_MODEL 1024
#define NHEAD   16
#define DK      64
#define BATCH   2
#define SEQ     2048
#define MROWS   (BATCH * SEQ)   // 4096

struct U4x2 { uint4 a, b; };

static __device__ __forceinline__ unsigned short f2bf(float f) {
    unsigned u = __builtin_bit_cast(unsigned, f);
    u += 0x7fffu + ((u >> 16) & 1u);          // round-to-nearest-even
    return (unsigned short)(u >> 16);
}

// Load one 16x32 bf16 WMMA fragment slice for this lane: two b128 chunks
// at element offsets 0 and +16 from p (p already includes row*stride + k0 + kb).
static __device__ __forceinline__ v16bf load_frag(const unsigned short* p) {
    U4x2 t;
    t.a = *(const uint4*)(p);
    t.b = *(const uint4*)(p + 16);
    return __builtin_bit_cast(v16bf, t);
}

// ---------------------------------------------------------------- cvt x -> bf16
__global__ __launch_bounds__(256) void cvt_x_kernel(const float* __restrict__ x,
                                                    unsigned short* __restrict__ xb,
                                                    int n4) {
    int i = blockIdx.x * blockDim.x + threadIdx.x;
    if (i >= n4) return;
    float4 v = ((const float4*)x)[i];
    ushort4 o;
    o.x = f2bf(v.x); o.y = f2bf(v.y); o.z = f2bf(v.z); o.w = f2bf(v.w);
    ((ushort4*)xb)[i] = o;
}

// ------------------------------------------- transpose W[K,N] -> Wt[N,K] bf16
__global__ __launch_bounds__(256) void wt_kernel(const float* __restrict__ W0,
                                                 const float* __restrict__ W1,
                                                 const float* __restrict__ W2,
                                                 const float* __restrict__ W3,
                                                 unsigned short* __restrict__ T0,
                                                 unsigned short* __restrict__ T1,
                                                 unsigned short* __restrict__ T2,
                                                 unsigned short* __restrict__ T3) {
    __shared__ float tile[32][33];
    const float* W = (blockIdx.z == 0) ? W0 : (blockIdx.z == 1) ? W1
                   : (blockIdx.z == 2) ? W2 : W3;
    unsigned short* T = (blockIdx.z == 0) ? T0 : (blockIdx.z == 1) ? T1
                      : (blockIdx.z == 2) ? T2 : T3;
    int x = blockIdx.x * 32 + threadIdx.x;          // n
#pragma unroll
    for (int j = 0; j < 4; ++j) {
        int y = blockIdx.y * 32 + threadIdx.y + j * 8;   // k
        tile[threadIdx.y + j * 8][threadIdx.x] = W[(size_t)y * D_MODEL + x];
    }
    __syncthreads();
    int x2 = blockIdx.y * 32 + threadIdx.x;         // k
#pragma unroll
    for (int j = 0; j < 4; ++j) {
        int y2 = blockIdx.x * 32 + threadIdx.y + j * 8;  // n
        T[(size_t)y2 * D_MODEL + x2] = f2bf(tile[threadIdx.x][threadIdx.y + j * 8]);
    }
}

// ------------------------------------------------------------------ bf16 GEMM
// A [M,K] bf16 row-major, Bt [N,K] bf16 row-major (pre-transposed weight).
// MODE 0: Q bf16 [B,H,S,Dk]   MODE 1: K bf16 [B,H,S,Dk]
// MODE 2: V bf16 [B,H,Dk,S]   MODE 3: out f32 [M,N]
// MODE is a compile-time constant so each instantiation has a straight-line
// epilogue (no per-element SALU branch ladders).
template <int MODE>
__global__ __launch_bounds__(256) void gemm_kernel(const unsigned short* __restrict__ A,
                                                   const unsigned short* __restrict__ Bt,
                                                   const float* __restrict__ bias,
                                                   void* __restrict__ Out) {
    const int K = D_MODEL, N = D_MODEL;
    int lane = threadIdx.x & 31;
    int wave = threadIdx.x >> 5;
    int wm = wave & 3, wn = wave >> 2;
    int rowBase = blockIdx.y * 128 + wm * 32;
    int colBase = blockIdx.x * 128 + wn * 64;
    int r    = lane & 15;
    int half = lane >> 4;
    int kb   = half * 8;

    v8f acc[2][4];
#pragma unroll
    for (int i = 0; i < 2; ++i)
#pragma unroll
        for (int j = 0; j < 4; ++j) acc[i][j] = (v8f){};

    for (int k0 = 0; k0 < K; k0 += 32) {
        v16bf af[2], bfx[4];
#pragma unroll
        for (int i = 0; i < 2; ++i) {
            const unsigned short* ap = A + (size_t)(rowBase + i * 16 + r) * K + k0 + kb;
            af[i] = load_frag(ap);
            __builtin_prefetch(ap + 32, 0, 0);   // next k-tile -> global_prefetch_b8
        }
#pragma unroll
        for (int j = 0; j < 4; ++j) {
            const unsigned short* bp = Bt + (size_t)(colBase + j * 16 + r) * K + k0 + kb;
            bfx[j] = load_frag(bp);
            __builtin_prefetch(bp + 32, 0, 0);
        }
#pragma unroll
        for (int i = 0; i < 2; ++i)
#pragma unroll
            for (int j = 0; j < 4; ++j)
                acc[i][j] = __builtin_amdgcn_wmma_f32_16x16x32_bf16(
                    false, af[i], false, bfx[j], (short)0, acc[i][j], false, false);
    }

#pragma unroll
    for (int i = 0; i < 2; ++i) {
#pragma unroll
        for (int j = 0; j < 4; ++j) {
#pragma unroll
            for (int rr = 0; rr < 8; ++rr) {
                int row = rowBase + i * 16 + rr + 8 * half;
                int col = colBase + j * 16 + (lane & 15);
                float v = acc[i][j][rr] + bias[col];
                if constexpr (MODE == 3) {
                    ((float*)Out)[(size_t)row * N + col] = v;
                } else {
                    int b = row >> 11, s = row & (SEQ - 1);
                    int h = col >> 6, d = col & (DK - 1);
                    unsigned short bv = f2bf(v);
                    if constexpr (MODE == 2) {
                        ((unsigned short*)Out)[((size_t)(b * NHEAD + h) * DK + d) * SEQ + s] = bv;
                    } else {
                        ((unsigned short*)Out)[((size_t)(b * NHEAD + h) * SEQ + s) * DK + d] = bv;
                    }
                }
            }
        }
    }
}

// -------------------------------------------------------- flash attention core
// Qb,Kb: [B,H,S,Dk] bf16.  Vt: [B,H,Dk,S] bf16.  Ob: [B,S,D] bf16.
__global__ __launch_bounds__(256) void attn_kernel(const unsigned short* __restrict__ Qb,
                                                   const unsigned short* __restrict__ Kb,
                                                   const unsigned short* __restrict__ Vt,
                                                   unsigned short* __restrict__ Ob) {
    __shared__ __align__(16) unsigned short Plds[8 * 16 * 64]; // 16KB: per-wave P tile
    int lane = threadIdx.x & 31;
    int wave = threadIdx.x >> 5;
    int bh = blockIdx.y;
    int qBase = blockIdx.x * 128 + wave * 16;
    int r = lane & 15, half = lane >> 4, kb = half * 8;
    const float scale = 0.125f; // 1/sqrt(64)

    const unsigned short* Qp = Qb + ((size_t)bh * SEQ + qBase) * DK;
    const unsigned short* Kp = Kb + (size_t)bh * SEQ * DK;
    const unsigned short* Vp = Vt + (size_t)bh * DK * SEQ;
    unsigned short* Pw = Plds + wave * 16 * 64;

    // Q fragments (16 rows x 64 K) live in registers for the whole pass.
    v16bf qf[2];
#pragma unroll
    for (int j = 0; j < 2; ++j)
        qf[j] = load_frag(Qp + (size_t)r * DK + j * 32 + kb);

    v8f o[4];
#pragma unroll
    for (int t = 0; t < 4; ++t) o[t] = (v8f){};
    float mrow[8], lrow[8];
#pragma unroll
    for (int rr = 0; rr < 8; ++rr) { mrow[rr] = -1e30f; lrow[rr] = 0.0f; }

    for (int kv = 0; kv < SEQ; kv += 64) {
        // ---- scores: S(16x64) = Q(16x64) x K^T, 8 WMMAs
        v8f sc[4];
#pragma unroll
        for (int t = 0; t < 4; ++t) sc[t] = (v8f){};
#pragma unroll
        for (int j = 0; j < 2; ++j) {
#pragma unroll
            for (int t = 0; t < 4; ++t) {
                const unsigned short* kp =
                    Kp + (size_t)(kv + t * 16 + r) * DK + j * 32 + kb;
                v16bf kf = load_frag(kp);
                __builtin_prefetch(kp + 64 * DK, 0, 0);  // next kv tile
                sc[t] = __builtin_amdgcn_wmma_f32_16x16x32_bf16(
                    false, qf[j], false, kf, (short)0, sc[t], false, false);
            }
        }

        // ---- online softmax (rows r+8*half live in one 16-lane half)
        float nm[8];
#pragma unroll
        for (int rr = 0; rr < 8; ++rr) {
            float m = sc[0][rr];
#pragma unroll
            for (int t = 1; t < 4; ++t) m = fmaxf(m, sc[t][rr]);
            for (int msk = 1; msk < 16; msk <<= 1)
                m = fmaxf(m, __shfl_xor(m, msk, 32));
            nm[rr] = fmaxf(mrow[rr], m * scale);
        }
#pragma unroll
        for (int rr = 0; rr < 8; ++rr) {
            float alpha = __expf(mrow[rr] - nm[rr]);
            mrow[rr] = nm[rr];
            float rs = 0.0f;
#pragma unroll
            for (int t = 0; t < 4; ++t) {
                float p = __expf(sc[t][rr] * scale - nm[rr]);
                rs += p;
                Pw[(rr + 8 * half) * 64 + t * 16 + (lane & 15)] = f2bf(p);
            }
            for (int msk = 1; msk < 16; msk <<= 1)
                rs += __shfl_xor(rs, msk, 32);
            lrow[rr] = lrow[rr] * alpha + rs;
#pragma unroll
            for (int t = 0; t < 4; ++t) o[t][rr] *= alpha;
        }

        // ---- O += P x V, 8 WMMAs (P re-read from LDS in A layout; same-wave
        //      DS ops are in order, so no barrier required)
#pragma unroll
        for (int j = 0; j < 2; ++j) {
            v16bf pf = load_frag(Pw + (size_t)r * 64 + j * 32 + kb);
#pragma unroll
            for (int t = 0; t < 4; ++t) {
                const unsigned short* vp =
                    Vp + (size_t)(t * 16 + r) * SEQ + kv + j * 32 + kb;
                v16bf vf = load_frag(vp);
                __builtin_prefetch(vp + 64, 0, 0);       // next kv tile
                o[t] = __builtin_amdgcn_wmma_f32_16x16x32_bf16(
                    false, pf, false, vf, (short)0, o[t], false, false);
            }
        }
    }

    // ---- normalize + write [B,S,D] bf16
    int b = bh >> 4, h = bh & (NHEAD - 1);
#pragma unroll
    for (int rr = 0; rr < 8; ++rr) {
        float inv = 1.0f / lrow[rr];
        int s = qBase + rr + 8 * half;
#pragma unroll
        for (int t = 0; t < 4; ++t) {
            int d = t * 16 + (lane & 15);
            Ob[((size_t)(b * SEQ + s)) * D_MODEL + h * DK + d] = f2bf(o[t][rr] * inv);
        }
    }
}

// --------------------------------------------------------------------- launch
extern "C" void kernel_launch(void* const* d_in, const int* in_sizes, int n_in,
                              void* d_out, int out_size, void* d_ws, size_t ws_size,
                              hipStream_t stream) {
    const float* x  = (const float*)d_in[0];
    const float* Wq = (const float*)d_in[1];
    const float* bq = (const float*)d_in[2];
    const float* Wk = (const float*)d_in[3];
    const float* bk = (const float*)d_in[4];
    const float* Wv = (const float*)d_in[5];
    const float* bv = (const float*)d_in[6];
    const float* Wo = (const float*)d_in[7];
    const float* bo = (const float*)d_in[8];
    float* out = (float*)d_out;

    char* ws = (char*)d_ws;
    const size_t MB = 1024ull * 1024ull;
    unsigned short* xb  = (unsigned short*)(ws + 0);        // 8 MiB  [4096,1024] bf16
    unsigned short* Wtq = (unsigned short*)(ws + 8 * MB);   // 2 MiB  [N,K] bf16
    unsigned short* Wtk = (unsigned short*)(ws + 10 * MB);
    unsigned short* Wtv = (unsigned short*)(ws + 12 * MB);
    unsigned short* Wto = (unsigned short*)(ws + 14 * MB);
    unsigned short* Qb  = (unsigned short*)(ws + 16 * MB);  // 8 MiB  [B,H,S,Dk]
    unsigned short* Kb  = (unsigned short*)(ws + 24 * MB);  // 8 MiB  [B,H,S,Dk]
    unsigned short* Vt  = (unsigned short*)(ws + 32 * MB);  // 8 MiB  [B,H,Dk,S]
    unsigned short* Ob  = (unsigned short*)(ws + 40 * MB);  // 8 MiB  [B,S,D]

    cvt_x_kernel<<<dim3((MROWS * D_MODEL / 4) / 256), dim3(256), 0, stream>>>(
        x, xb, MROWS * D_MODEL / 4);
    wt_kernel<<<dim3(32, 32, 4), dim3(32, 8), 0, stream>>>(
        Wq, Wk, Wv, Wo, Wtq, Wtk, Wtv, Wto);

    gemm_kernel<0><<<dim3(8, 32), dim3(256), 0, stream>>>(xb, Wtq, bq, (void*)Qb);
    gemm_kernel<1><<<dim3(8, 32), dim3(256), 0, stream>>>(xb, Wtk, bk, (void*)Kb);
    gemm_kernel<2><<<dim3(8, 32), dim3(256), 0, stream>>>(xb, Wtv, bv, (void*)Vt);

    attn_kernel<<<dim3(SEQ / 128, BATCH * NHEAD), dim3(256), 0, stream>>>(Qb, Kb, Vt, Ob);

    gemm_kernel<3><<<dim3(8, 32), dim3(256), 0, stream>>>(Ob, Wto, bo, (void*)out);
}
